// ShaDowGCN_55490977465225
// MI455X (gfx1250) — matrix operand order repaired and checked by reading, not compiled, not run on hardware
//
#include <hip/hip_runtime.h>
#include <hip/hip_bf16.h>

typedef __attribute__((ext_vector_type(16))) __bf16 v16bf;
typedef __attribute__((ext_vector_type(4)))  __bf16 v4bf;
typedef __attribute__((ext_vector_type(8)))  float  v8f;

#define DD      128
#define OUTC    9
#define NLAYERS 4
#define ROWS    32   // nodes per GEMM block

__device__ __forceinline__ void atomAddF(float* p, float v) {
    __hip_atomic_fetch_add(p, v, __ATOMIC_RELAXED, __HIP_MEMORY_SCOPE_AGENT);
}

// ---------------- weight conversion + WMMA-fragment swizzle ----------------
// For B fragments (32x16 bf16 tiles) this code's per-lane layout (contiguous k, not AMD's table order) is:
//   element pair j of lane (col, half): W[kk + half*16 + 2j][col], W[kk + half*16 + 2j + 1][col]
// We store those 16 values contiguously so the GEMM loads one aligned v16bf per fragment:
//   dest = ((k>>5)*2*DD + col*2 + half)*16 + 2*((k&15)>>1) + (k&1)

__global__ void k_convert_w(const float* __restrict__ Wa, const float* __restrict__ Wr,
                            __bf16* __restrict__ WaS, __bf16* __restrict__ WrS, int n) {
    int i = blockIdx.x * blockDim.x + threadIdx.x;
    if (i >= n) return;
    int layer = i / (DD * DD);
    int idx   = i - layer * DD * DD;
    int k     = idx / DD;
    int col   = idx - k * DD;
    int kk    = k >> 5;
    int kq    = k & 31;
    int half  = kq >> 4;
    int j     = (kq & 15) >> 1;
    int t     = kq & 1;
    size_t dest = (size_t)layer * DD * DD +
                  ((size_t)(kk * 2 * DD + col * 2 + half) << 4) + 2 * j + t;
    WaS[dest] = (__bf16)Wa[i];
    WrS[dest] = (__bf16)Wr[i];
}

__global__ void k_copy(const float* __restrict__ src, float* __restrict__ dst, int n) {
    int i = blockIdx.x * blockDim.x + threadIdx.x;
    if (i < n) dst[i] = src[i];
}

__global__ void k_deg(const int* __restrict__ dstIdx, float* __restrict__ deg, int E) {
    int e = blockIdx.x * blockDim.x + threadIdx.x;
    if (e < E) atomAddF(&deg[dstIdx[e]], 1.0f);
}

// ---------------- edge scatter (mean-aggregation numerator) ----------------

__global__ __launch_bounds__(256) void k_scatter(const float* __restrict__ h,
                                                 const int* __restrict__ src,
                                                 const int* __restrict__ dst,
                                                 float* __restrict__ agg, int E) {
    int t  = blockIdx.x * blockDim.x + threadIdx.x;
    int e  = t >> 5;
    int c4 = (t & 31) << 2;
    if (e < E) {
        int s = src[e];
        int d = dst[e];
        const float4 v = *reinterpret_cast<const float4*>(h + (size_t)s * DD + c4);
        float* a = agg + (size_t)d * DD + c4;
        atomAddF(a + 0, v.x);
        atomAddF(a + 1, v.y);
        atomAddF(a + 2, v.z);
        atomAddF(a + 3, v.w);
    }
}

// ---------------- WMMA SAGEConv layer: h = relu(mean@Wa + ba + h@Wr) in place ----------------

__global__ __launch_bounds__(256) void k_gemm(float* __restrict__ h,
                                              const float* __restrict__ agg,
                                              const float* __restrict__ deg,
                                              const __bf16* __restrict__ WaS,
                                              const __bf16* __restrict__ WrS,
                                              const float* __restrict__ ba) {
    __shared__ __bf16 sA[2][ROWS][DD];   // [0]=mean rows, [1]=root rows (bf16)

    const int tid   = threadIdx.x;
    const int node0 = blockIdx.x * ROWS;

    // stage 32 node rows of mean + root features into LDS (float4 loads, v4bf stores)
    for (int idx = tid; idx < ROWS * (DD / 4); idx += 256) {
        int r  = idx >> 5;               // DD/4 == 32 chunks per row
        int k4 = (idx & 31) << 2;
        int n  = node0 + r;
        float dg = deg[n];
        dg = dg > 1.0f ? dg : 1.0f;
        float inv = 1.0f / dg;
        const float4 va = *reinterpret_cast<const float4*>(agg + (size_t)n * DD + k4);
        const float4 vh = *reinterpret_cast<const float4*>(h   + (size_t)n * DD + k4);
        *reinterpret_cast<v4bf*>(&sA[0][r][k4]) =
            (v4bf){(__bf16)(va.x * inv), (__bf16)(va.y * inv),
                   (__bf16)(va.z * inv), (__bf16)(va.w * inv)};
        *reinterpret_cast<v4bf*>(&sA[1][r][k4]) =
            (v4bf){(__bf16)vh.x, (__bf16)vh.y, (__bf16)vh.z, (__bf16)vh.w};
    }
    __syncthreads();

    const int wave = tid >> 5;          // 8 waves -> 8 column tiles of 16
    const int lane = tid & 31;
    const int half = lane >> 4;
    const int ln   = lane & 15;
    const int col0 = wave * 16;
    const int col  = col0 + ln;

    const float bias = ba[col];

    v8f acc[2];
#pragma unroll
    for (int rt = 0; rt < 2; ++rt)
#pragma unroll
        for (int i = 0; i < 8; ++i) acc[rt][i] = bias;

#pragma unroll
    for (int kk = 0; kk < DD; kk += 32) {
        // pre-swizzled B fragments: one aligned 32-byte v16bf load each
        const size_t fragOff = (size_t)(((kk >> 5) * 2 * DD + col * 2 + half) << 4);
        const v16bf bwa = *reinterpret_cast<const v16bf*>(WaS + fragOff);
        const v16bf bwr = *reinterpret_cast<const v16bf*>(WrS + fragOff);

#pragma unroll
        for (int rt = 0; rt < 2; ++rt) {
            int m = rt * 16 + ln;
            // A fragments (16x32, 16-bit), contiguous k order matching B: VGPR j: K = half*16 + 2*j, +1
            v16bf am, ar;
#pragma unroll
            for (int j = 0; j < 8; ++j) {
                int kp = kk + half * 16 + 2 * j;
                am[2 * j]     = sA[0][m][kp];
                am[2 * j + 1] = sA[0][m][kp + 1];
                ar[2 * j]     = sA[1][m][kp];
                ar[2 * j + 1] = sA[1][m][kp + 1];
            }
            acc[rt] = __builtin_amdgcn_wmma_f32_16x16x32_bf16(
                false, am, false, bwa, (short)0, acc[rt], false, false);
            acc[rt] = __builtin_amdgcn_wmma_f32_16x16x32_bf16(
                false, ar, false, bwr, (short)0, acc[rt], false, false);
        }
    }

    // C/D layout: VGPR r holds M = r + 8*half, N = ln ; fuse relu, write in place
#pragma unroll
    for (int rt = 0; rt < 2; ++rt)
#pragma unroll
        for (int r = 0; r < 8; ++r) {
            int m = rt * 16 + half * 8 + r;
            float v = acc[rt][r];
            h[(size_t)(node0 + m) * DD + col] = v > 0.0f ? v : 0.0f;
        }
}

// ---------------- global mean pool scatter ----------------

__global__ __launch_bounds__(256) void k_pool(const float* __restrict__ h,
                                              const int* __restrict__ batch,
                                              float* __restrict__ psum,
                                              float* __restrict__ cnt, int N) {
    int t  = blockIdx.x * blockDim.x + threadIdx.x;
    int n  = t >> 5;
    int c4 = (t & 31) << 2;
    if (n < N) {
        int b = batch[n];
        const float4 v = *reinterpret_cast<const float4*>(h + (size_t)n * DD + c4);
        float* p = psum + (size_t)b * DD + c4;
        atomAddF(p + 0, v.x);
        atomAddF(p + 1, v.y);
        atomAddF(p + 2, v.z);
        atomAddF(p + 3, v.w);
        if (c4 == 0) atomAddF(&cnt[b], 1.0f);
    }
}

// ---------------- head: concat(h[root], pooled) @ W_lin + b_lin, log_softmax ----------------

__global__ void k_head(const float* __restrict__ h, const float* __restrict__ psum,
                       const float* __restrict__ cnt, const int* __restrict__ root,
                       const float* __restrict__ Wl, const float* __restrict__ bl,
                       float* __restrict__ out, int B) {
    int b = blockIdx.x * blockDim.x + threadIdx.x;
    if (b >= B) return;
    int r = root[b];
    float c = cnt[b];
    c = c > 1.0f ? c : 1.0f;
    float inv = 1.0f / c;

    float lg[OUTC];
#pragma unroll
    for (int o = 0; o < OUTC; ++o) lg[o] = bl[o];

    const float* hr = h + (size_t)r * DD;
    const float* ps = psum + (size_t)b * DD;
    for (int k = 0; k < DD; ++k) {
        float f = hr[k];
        const float* w = Wl + (size_t)k * OUTC;
#pragma unroll
        for (int o = 0; o < OUTC; ++o) lg[o] += f * w[o];
    }
    for (int k = 0; k < DD; ++k) {
        float f = ps[k] * inv;
        const float* w = Wl + (size_t)(DD + k) * OUTC;
#pragma unroll
        for (int o = 0; o < OUTC; ++o) lg[o] += f * w[o];
    }

    float m = lg[0];
#pragma unroll
    for (int o = 1; o < OUTC; ++o) m = lg[o] > m ? lg[o] : m;
    float s = 0.0f;
#pragma unroll
    for (int o = 0; o < OUTC; ++o) s += __expf(lg[o] - m);
    float ls = __logf(s);
#pragma unroll
    for (int o = 0; o < OUTC; ++o) out[(size_t)b * OUTC + o] = lg[o] - m - ls;
}

// ---------------- launch ----------------

extern "C" void kernel_launch(void* const* d_in, const int* in_sizes, int n_in,
                              void* d_out, int out_size, void* d_ws, size_t ws_size,
                              hipStream_t stream) {
    const float* x      = (const float*)d_in[0];
    const int*   ei     = (const int*)d_in[1];
    const int*   batch  = (const int*)d_in[2];
    const int*   root   = (const int*)d_in[3];
    const float* W_agg  = (const float*)d_in[4];
    const float* W_root = (const float*)d_in[5];
    const float* b_agg  = (const float*)d_in[6];
    const float* W_lin  = (const float*)d_in[7];
    const float* b_lin  = (const float*)d_in[8];

    const int N = in_sizes[0] / DD;
    const int E = in_sizes[1] / 2;
    const int B = in_sizes[3];
    const size_t ND = (size_t)N * DD;

    // workspace carve-up
    float*  h    = (float*)d_ws;
    float*  agg  = h + ND;
    float*  deg  = agg + ND;
    float*  psum = deg + N;
    float*  cnt  = psum + (size_t)B * DD;
    __bf16* WaS  = (__bf16*)(cnt + B);
    __bf16* WrS  = WaS + (size_t)NLAYERS * DD * DD;

    const int* srcI = ei;
    const int* dstI = ei + E;

    // bf16 weights, pre-swizzled into WMMA B-fragment order
    const int nw = NLAYERS * DD * DD;
    k_convert_w<<<(nw + 255) / 256, 256, 0, stream>>>(W_agg, W_root, WaS, WrS, nw);

    // degrees (layer-invariant)
    hipMemsetAsync(deg, 0, (size_t)N * sizeof(float), stream);
    k_deg<<<(E + 255) / 256, 256, 0, stream>>>(dstI, deg, E);

    // h = x
    k_copy<<<(int)((ND + 255) / 256), 256, 0, stream>>>(x, h, (int)ND);

    const int scatterBlocks = (E * 32 + 255) / 256;
    for (int l = 0; l < NLAYERS; ++l) {
        hipMemsetAsync(agg, 0, ND * sizeof(float), stream);
        k_scatter<<<scatterBlocks, 256, 0, stream>>>(h, srcI, dstI, agg, E);
        k_gemm<<<N / ROWS, 256, 0, stream>>>(h, agg, deg,
                                             WaS + (size_t)l * DD * DD,
                                             WrS + (size_t)l * DD * DD,
                                             b_agg + (size_t)l * DD);
    }

    // pooling + head
    hipMemsetAsync(psum, 0, (size_t)B * DD * sizeof(float), stream);
    hipMemsetAsync(cnt, 0, (size_t)B * sizeof(float), stream);
    k_pool<<<(N * 32 + 255) / 256, 256, 0, stream>>>(h, batch, psum, cnt, N);
    k_head<<<(B + 255) / 256, 256, 0, stream>>>(h, psum, cnt, root, W_lin, b_lin,
                                                (float*)d_out, B);
}
